// GraphFusion_51264729645564
// MI455X (gfx1250) — matrix operand order repaired and checked by reading, not compiled, not run on hardware
//
#include <hip/hip_runtime.h>
#include <hip/hip_bf16.h>
#include <stdint.h>

// Problem constants (match reference): B=4, N=256, D=128, E=64, L=3
#define BDIM   256
#define NNODES 1024   // B*N
#define DD     128    // D
#define EE     64     // E
#define H2     256    // 2*D
#define NJ     256    // N (neighbors)
#define NLAYER 3

typedef __bf16 v16bf_t __attribute__((ext_vector_type(16)));
typedef float  v8f_t   __attribute__((ext_vector_type(8)));

union Frag { uint32_t u[8]; v16bf_t v; };

__device__ __forceinline__ uint32_t f2bf1(float f) {
  uint32_t u = __builtin_bit_cast(uint32_t, f);
  return ((u + 0x7FFFu + ((u >> 16) & 1u)) >> 16) & 0xFFFFu;  // RNE bf16
}
__device__ __forceinline__ uint32_t packbf(float lo, float hi) {
  return f2bf1(lo) | (f2bf1(hi) << 16);
}
__device__ __forceinline__ uint16_t f2bf16(float f) { return (uint16_t)f2bf1(f); }

#define WMMA_BF16(Afrag, Bfrag, Cacc) \
  __builtin_amdgcn_wmma_f32_16x16x32_bf16(false, (Afrag).v, false, (Bfrag).v, (short)0, (Cacc), false, false)

// ---------------------------------------------------------------------------
// Kernel 1: per-node precompute  xi_top = x@ew1[:D], xj_bot = x@ew1[D:],
//           xm = x@mw1[:D]   (tiny GEMVs, fp32)
// ---------------------------------------------------------------------------
__global__ __launch_bounds__(BDIM) void precompute_kernel(
    const float* __restrict__ x, const float* __restrict__ ew1g,
    const float* __restrict__ mw1g,
    float* __restrict__ xi_top, float* __restrict__ xj_bot,
    float* __restrict__ xm)
{
  __shared__ float xr[DD];
  const int tid = threadIdx.x, node = blockIdx.x;
  if (tid < DD) xr[tid] = x[(size_t)node * DD + tid];
  __syncthreads();
  {
    float acc = 0.f;
    for (int d = 0; d < DD; ++d) acc += xr[d] * mw1g[d * H2 + tid];
    xm[(size_t)node * H2 + tid] = acc;
  }
  if (tid < EE) {
    float acc = 0.f;
    for (int d = 0; d < DD; ++d) acc += xr[d] * ew1g[d * EE + tid];
    xi_top[(size_t)node * EE + tid] = acc;
  } else if (tid < 2 * EE) {
    const int e = tid - EE;
    float acc = 0.f;
    for (int d = 0; d < DD; ++d) acc += xr[d] * ew1g[(DD + d) * EE + e];
    xj_bot[(size_t)node * EE + e] = acc;
  }
}

// ---------------------------------------------------------------------------
// Kernel 2: fused edge-network + message GEMMs with WMMA bf16, per (b,i).
//   eh[j]        = relu(xi_top[i] + xj_bot[j] + eb1)        (bf16 in LDS)
//   edge_feat[j] = eh @ ew2 + eb2                           (WMMA, bf16 back)
//   mh[j]        = relu(xm[i] + edge_feat @ mw_bot + mb1)   (WMMA, regs)
//   s[i]         = sum_j adj[i,j] * mh[j]  ;  asum = sum_j adj[i,j]
// Each wave owns row tiles {wave, wave+8}; B fragments shared across both.
// ---------------------------------------------------------------------------
#define SMEM_MAIN 77312

__global__ __launch_bounds__(BDIM) void gnn_edge_msg_kernel(
    const float* __restrict__ xi_top, const float* __restrict__ xj_bot,
    const float* __restrict__ xm,     const float* __restrict__ adj,
    const float* __restrict__ ew2g,   const float* __restrict__ eb1g,
    const float* __restrict__ eb2g,   const float* __restrict__ mw1g,
    const float* __restrict__ mb1g,
    float* __restrict__ s_buf, float* __restrict__ asum_buf)
{
  extern __shared__ char smem[];
  uint32_t* mwbot_w = (uint32_t*)smem;              // [256n][32w]  mw_bot^T bf16x2
  uint32_t* ewt_w   = mwbot_w + 256 * 32;           // [64n][32w]   ew2^T   bf16x2
  uint16_t* e_h     = (uint16_t*)(ewt_w + 64 * 32); // [256j][64e]  eh -> edge_feat
  uint32_t* e_w     = (uint32_t*)e_h;               // word view    [256j][32w]
  float* a_row = (float*)(e_h + 256 * 64);          // [256]
  float* s_lds = a_row + 256;                       // [256]
  float* c1    = s_lds + 256;                       // [64]  xi_top + eb1
  float* eb2s  = c1 + 64;                           // [64]
  float* c2    = eb2s + 64;                         // [256] xm_i + mb1

  const int tid  = threadIdx.x;
  const int node = blockIdx.x;      // b*N + i
  const int b    = node >> 8;

  // ---- stage weights / biases / adjacency (all coalesced) ----
  for (int it = 0; it < 32; ++it) {               // mw_bot^T : 8192 words
    const int idx = tid + it * BDIM;
    const int n = idx & 255, w = idx >> 8;
    mwbot_w[n * 32 + w] =
        packbf(mw1g[(DD + 2 * w) * H2 + n], mw1g[(DD + 2 * w + 1) * H2 + n]);
  }
  for (int it = 0; it < 8; ++it) {                // ew2^T : 2048 words
    const int idx = tid + it * BDIM;
    const int n = idx & 63, w = idx >> 6;
    ewt_w[n * 32 + w] = packbf(ew2g[(2 * w) * EE + n], ew2g[(2 * w + 1) * EE + n]);
  }
  a_row[tid] = adj[(size_t)node * NJ + tid];
  s_lds[tid] = 0.f;
  if (tid < EE) { c1[tid] = xi_top[(size_t)node * EE + tid] + eb1g[tid]; eb2s[tid] = eb2g[tid]; }
  c2[tid] = xm[(size_t)node * H2 + tid] + mb1g[tid];
  __syncthreads();

  // ---- eh: pair-indexed, coalesced float2 reads of xj_bot ----
  {
    const float2* xjb2 = (const float2*)(xj_bot + (size_t)b * NJ * EE);
    for (int it = 0; it < 8; ++it) {
      const int idx = tid + it * BDIM;            // pair index 0..2047
      const int j = idx >> 5, w = idx & 31;
      const float2 p = xjb2[idx];
      float v0 = c1[2 * w]     + p.x;
      float v1 = c1[2 * w + 1] + p.y;
      v0 = v0 > 0.f ? v0 : 0.f;
      v1 = v1 > 0.f ? v1 : 0.f;
      e_w[j * 32 + w] = packbf(v0, v1);
    }
  }
  __syncthreads();

  // ---- WMMA phase ----
  const int wave = tid >> 5;
  const int lane = tid & 31;
  const int lo = lane & 15;     // column within tile (B/C/D layout)
  const int hi = lane >> 4;     // half select
  const int j0a = wave * 16;
  const int j0b = (wave + 8) * 16;

  // A fragments of eh for both row tiles (contiguous k = 16*hi + s order, matching B)
  Frag A[2][2];
  #pragma unroll
  for (int rr = 0; rr < 2; ++rr) {
    const int j0 = rr ? j0b : j0a;
    #pragma unroll
    for (int kt = 0; kt < 2; ++kt)
      #pragma unroll
      for (int v = 0; v < 8; ++v) {
        const int kw = kt * 16 + hi * 8 + v;
        A[rr][kt].u[v] = e_w[(j0 + lo) * 32 + kw];
      }
  }

  // GEMM1: edge_feat = eh @ ew2 + eb2 -> bf16 back into e_h (own rows only)
  #pragma unroll
  for (int nt = 0; nt < 4; ++nt) {
    Frag B0, B1;
    #pragma unroll
    for (int v = 0; v < 8; ++v) {
      B0.u[v] = ewt_w[(nt * 16 + lo) * 32 + hi * 8 + v];
      B1.u[v] = ewt_w[(nt * 16 + lo) * 32 + 16 + hi * 8 + v];
    }
    v8f_t acc0 = {}, acc1 = {};
    acc0 = WMMA_BF16(A[0][0], B0, acc0);
    acc1 = WMMA_BF16(A[1][0], B0, acc1);
    acc0 = WMMA_BF16(A[0][1], B1, acc0);
    acc1 = WMMA_BF16(A[1][1], B1, acc1);
    const float bias = eb2s[nt * 16 + lo];
    #pragma unroll
    for (int r = 0; r < 8; ++r) {
      e_h[(j0a + r + hi * 8) * 64 + nt * 16 + lo] = f2bf16(acc0[r] + bias);
      e_h[(j0b + r + hi * 8) * 64 + nt * 16 + lo] = f2bf16(acc1[r] + bias);
    }
  }

  // A fragments of edge_feat (per-wave DS ordering makes stores visible)
  #pragma unroll
  for (int rr = 0; rr < 2; ++rr) {
    const int j0 = rr ? j0b : j0a;
    #pragma unroll
    for (int kt = 0; kt < 2; ++kt)
      #pragma unroll
      for (int v = 0; v < 8; ++v) {
        const int kw = kt * 16 + hi * 8 + v;
        A[rr][kt].u[v] = e_w[(j0 + lo) * 32 + kw];
      }
  }

  float aw0[8], aw1[8];
  #pragma unroll
  for (int r = 0; r < 8; ++r) {
    aw0[r] = a_row[j0a + r + hi * 8];
    aw1[r] = a_row[j0b + r + hi * 8];
  }

  // GEMM2: mh = relu(c2 + edge_feat @ mw_bot), fused adjacency reduction
  #pragma unroll 4
  for (int nt = 0; nt < 16; ++nt) {
    Frag B0, B1;
    #pragma unroll
    for (int v = 0; v < 8; ++v) {
      B0.u[v] = mwbot_w[(nt * 16 + lo) * 32 + hi * 8 + v];
      B1.u[v] = mwbot_w[(nt * 16 + lo) * 32 + 16 + hi * 8 + v];
    }
    v8f_t acc0 = {}, acc1 = {};
    acc0 = WMMA_BF16(A[0][0], B0, acc0);
    acc1 = WMMA_BF16(A[1][0], B0, acc1);
    acc0 = WMMA_BF16(A[0][1], B1, acc0);
    acc1 = WMMA_BF16(A[1][1], B1, acc1);
    const float bias = c2[nt * 16 + lo];
    float partial = 0.f;
    #pragma unroll
    for (int r = 0; r < 8; ++r) {
      float m0 = acc0[r] + bias;
      float m1 = acc1[r] + bias;
      m0 = m0 > 0.f ? m0 : 0.f;
      m1 = m1 > 0.f ? m1 : 0.f;
      partial += aw0[r] * m0 + aw1[r] * m1;
    }
    partial += __shfl_xor(partial, 16, 32);   // merge lane halves (rows 0-7 / 8-15)
    if (lane < 16) atomicAdd(&s_lds[nt * 16 + lane], partial);
  }
  __syncthreads();

  s_buf[(size_t)node * H2 + tid] = s_lds[tid];
  if (tid == 0) {
    float a = 0.f;
    for (int j = 0; j < NJ; ++j) a += a_row[j];
    asum_buf[node] = a;
  }
}

// ---------------------------------------------------------------------------
// Kernel 3: messages = s@mw2 + asum*mb2; update MLP; residual + LayerNorm
// ---------------------------------------------------------------------------
__global__ __launch_bounds__(BDIM) void update_kernel(
    const float* __restrict__ x, const float* __restrict__ s_buf,
    const float* __restrict__ asum_buf,
    const float* __restrict__ mw2g, const float* __restrict__ mb2g,
    const float* __restrict__ uw1g, const float* __restrict__ ub1g,
    const float* __restrict__ uw2g, const float* __restrict__ ub2g,
    const float* __restrict__ lng,  const float* __restrict__ lnb,
    float* __restrict__ xout)
{
  __shared__ float xr[DD], msg[DD], sr[H2], uh[H2], red[DD];
  const int tid = threadIdx.x, node = blockIdx.x;
  if (tid < DD) xr[tid] = x[(size_t)node * DD + tid];
  sr[tid] = s_buf[(size_t)node * H2 + tid];
  __syncthreads();
  const float asum = asum_buf[node];
  if (tid < DD) {
    float acc = mb2g[tid] * asum;
    for (int h = 0; h < H2; ++h) acc += sr[h] * mw2g[h * DD + tid];
    msg[tid] = acc;
  }
  __syncthreads();
  {
    float acc = ub1g[tid];
    for (int k = 0; k < DD; ++k) acc += xr[k] * uw1g[k * H2 + tid];
    for (int k = 0; k < DD; ++k) acc += msg[k] * uw1g[(DD + k) * H2 + tid];
    uh[tid] = acc > 0.f ? acc : 0.f;
  }
  __syncthreads();
  float y = 0.f;
  if (tid < DD) {
    float acc = ub2g[tid];
    for (int k = 0; k < H2; ++k) acc += uh[k] * uw2g[k * DD + tid];
    y = xr[tid] + acc;
    red[tid] = y;
  }
  __syncthreads();
  for (int st = 64; st > 0; st >>= 1) {
    if (tid < st) red[tid] += red[tid + st];
    __syncthreads();
  }
  const float mu = red[0] * (1.f / DD);
  __syncthreads();
  if (tid < DD) { const float d0 = y - mu; red[tid] = d0 * d0; }
  __syncthreads();
  for (int st = 64; st > 0; st >>= 1) {
    if (tid < st) red[tid] += red[tid + st];
    __syncthreads();
  }
  const float var = red[0] * (1.f / DD);
  const float inv = rsqrtf(var + 1e-5f);
  if (tid < DD)
    xout[(size_t)node * DD + tid] = (y - mu) * inv * lng[tid] + lnb[tid];
}

// ---------------------------------------------------------------------------
extern "C" void kernel_launch(void* const* d_in, const int* in_sizes, int n_in,
                              void* d_out, int out_size, void* d_ws, size_t ws_size,
                              hipStream_t stream) {
  (void)in_sizes; (void)n_in; (void)out_size; (void)ws_size;
  const float* x_in = (const float*)d_in[0];
  const float* adj  = (const float*)d_in[1];
  const float* ew1  = (const float*)d_in[2];   // [3][256][64]
  const float* eb1  = (const float*)d_in[3];   // [3][64]
  const float* ew2  = (const float*)d_in[4];   // [3][64][64]
  const float* eb2  = (const float*)d_in[5];   // [3][64]
  const float* mw1  = (const float*)d_in[6];   // [3][192][256]
  const float* mb1  = (const float*)d_in[7];   // [3][256]
  const float* mw2  = (const float*)d_in[8];   // [3][256][128]
  const float* mb2  = (const float*)d_in[9];   // [3][128]
  const float* uw1  = (const float*)d_in[10];  // [3][256][256]
  const float* ub1  = (const float*)d_in[11];  // [3][256]
  const float* uw2  = (const float*)d_in[12];  // [3][256][128]
  const float* ub2  = (const float*)d_in[13];  // [3][128]
  const float* lng  = (const float*)d_in[14];  // [3][128]
  const float* lnb  = (const float*)d_in[15];  // [3][128]

  float* ws     = (float*)d_ws;
  float* xi_top = ws;                          // 1024*64
  float* xj_bot = xi_top + NNODES * EE;        // 1024*64
  float* xm     = xj_bot + NNODES * EE;        // 1024*256
  float* s_buf  = xm + NNODES * H2;            // 1024*256
  float* asum   = s_buf + NNODES * H2;         // 1024
  float* xb0    = asum + NNODES;               // 1024*128
  float* xb1    = xb0 + NNODES * DD;           // 1024*128

  for (int l = 0; l < NLAYER; ++l) {
    const float* xcur = (l == 0) ? x_in : ((l == 1) ? xb0 : xb1);
    float* xnext = (l == 0) ? xb0 : ((l == 1) ? xb1 : (float*)d_out);

    precompute_kernel<<<NNODES, BDIM, 0, stream>>>(
        xcur, ew1 + (size_t)l * 256 * 64, mw1 + (size_t)l * 192 * 256,
        xi_top, xj_bot, xm);

    gnn_edge_msg_kernel<<<NNODES, BDIM, SMEM_MAIN, stream>>>(
        xi_top, xj_bot, xm, adj,
        ew2 + (size_t)l * 64 * 64, eb1 + (size_t)l * 64, eb2 + (size_t)l * 64,
        mw1 + (size_t)l * 192 * 256, mb1 + (size_t)l * 256,
        s_buf, asum);

    update_kernel<<<NNODES, BDIM, 0, stream>>>(
        xcur, s_buf, asum,
        mw2 + (size_t)l * 256 * 128, mb2 + (size_t)l * 128,
        uw1 + (size_t)l * 256 * 256, ub1 + (size_t)l * 256,
        uw2 + (size_t)l * 256 * 128, ub2 + (size_t)l * 128,
        lng + (size_t)l * 128, lnb + (size_t)l * 128, xnext);
  }
}
